// Node_Routing_SelfAttentionLayer_8486855377191
// MI455X (gfx1250) — matrix operand order repaired and checked by reading, not compiled, not run on hardware
//
#include <hip/hip_runtime.h>
#include <hip/hip_bf16.h>

typedef __bf16 v16bf __attribute__((ext_vector_type(16)));
typedef __bf16 v8bf  __attribute__((ext_vector_type(8)));
typedef float  v8f   __attribute__((ext_vector_type(8)));

#define TOK   128
#define MDIM  128
#define HD    16
#define FFD   512
#define LDH   136   // padded bf16 row stride for 128-wide buffers (272B rows, 16B aligned)
#define LDH2  264   // padded bf16 row stride for 256-wide buffers (528B rows, 16B aligned)
#define WGSZ  256

// ---- LDS layout (bytes) ----
#define OFF_XF   0          // f32 residual stream  128*128*4 = 65536
#define OFF_XBF  65536      // bf16 x               128*136*2 = 34816
#define OFF_Q    100352     // bf16 Q token-major   34816
#define OFF_K    135168     // bf16 K token-major   34816
#define OFF_VT   169984     // bf16 V feature-major 34816
#define OFF_P    204800     // bf16 probs (per-wave 16x136 strips) 34816
#define OFF_AO   239616     // bf16 attn out token-major 34816
#define OFF_WT   274432     // bf16 staged transposed weight 34816
#define SMEM_TOT 309248
// FF aliases: w1T/w2T at OFF_Q (<=69632 over Q+K), hbuf at OFF_VT (<=69632 over VT+P)

union BF16x16 { v16bf v; v8bf h[2]; };

__device__ __forceinline__ v8f wmma_bf16(v16bf a, v16bf b, v8f c) {
  return __builtin_amdgcn_wmma_f32_16x16x32_bf16(false, a, false, b, (short)0, c, false, false);
}

__device__ __forceinline__ v8bf zero8() {
  v8bf z;
#pragma unroll
  for (int i = 0; i < 8; ++i) z[i] = (__bf16)0.0f;
  return z;
}

// A operand: row = lane&15; halves 0..7 hold K = kstart+8*lh .. +7, halves 8..15 hold K+16
__device__ __forceinline__ v16bf load_a(const __bf16* row, int kstart, int lh) {
  BF16x16 t;
  t.h[0] = *(const v8bf*)(row + kstart + lh * 8);
  t.h[1] = *(const v8bf*)(row + kstart + 16 + lh * 8);
  return t.v;
}
// B operand from transposed matrix BT[n][k]: col = lane&15; halves 0..7 hold K = kstart+8*lh .. +7, halves 8..15 hold K+16
__device__ __forceinline__ v16bf load_b(const __bf16* colrow, int kstart, int lh) {
  BF16x16 t;
  t.h[0] = *(const v8bf*)(colrow + kstart + lh * 8);
  t.h[1] = *(const v8bf*)(colrow + kstart + 16 + lh * 8);
  return t.v;
}

__device__ __forceinline__ float rsum16(float v) {
  v += __shfl_xor(v, 1); v += __shfl_xor(v, 2);
  v += __shfl_xor(v, 4); v += __shfl_xor(v, 8);
  return v;
}
__device__ __forceinline__ float rmax16(float v) {
  v = fmaxf(v, __shfl_xor(v, 1)); v = fmaxf(v, __shfl_xor(v, 2));
  v = fmaxf(v, __shfl_xor(v, 4)); v = fmaxf(v, __shfl_xor(v, 8));
  return v;
}

// stage 128x128 f32 weight W[k][n] -> bf16 transposed T[n*LDH + k]
__device__ __forceinline__ void stageT128(const float* W, __bf16* T, int tid) {
  for (int i = tid; i < MDIM * MDIM; i += WGSZ) {
    int k = i >> 7, n = i & 127;
    T[n * LDH + k] = (__bf16)W[k * MDIM + n];
  }
}

// One 16-token-row projection GEMM: dst[row*RS + col*CS] = bf16(Xbf @ Wt + bias)
// RS/CS are compile-time so stores are straight-line (no per-store branching).
template <int RS, int CS>
__device__ __forceinline__ void proj_gemm(const __bf16* Xsrc, const __bf16* Wt,
                                          const float* bias, __bf16* dst,
                                          int wave, int ln, int lh) {
  v16bf At[4];
#pragma unroll
  for (int kk = 0; kk < 4; ++kk)
    At[kk] = load_a(&Xsrc[(wave * 16 + ln) * LDH], kk * 32, lh);
#pragma unroll 1
  for (int nb = 0; nb < 8; ++nb) {
    float b = bias[nb * 16 + ln];
    v8f acc;
#pragma unroll
    for (int i = 0; i < 8; ++i) acc[i] = b;
    const __bf16* bptr = &Wt[(nb * 16 + ln) * LDH];
#pragma unroll
    for (int kk = 0; kk < 4; ++kk)
      acc = wmma_bf16(At[kk], load_b(bptr, kk * 32, lh), acc);
#pragma unroll
    for (int v = 0; v < 8; ++v) {
      int tok = wave * 16 + v + 8 * lh;
      dst[tok * RS + (nb * 16 + ln) * CS] = (__bf16)acc[v];
    }
  }
}

extern "C" __global__ void __launch_bounds__(WGSZ)
fused_node_attn_layer(const float* __restrict__ route,
                      const float* wq, const float* bq,
                      const float* wk, const float* bk,
                      const float* wv, const float* bv,
                      const float* wo, const float* bo,
                      const float* w1, const float* b1,
                      const float* w2, const float* b2,
                      const float* g1, const float* be1,
                      const float* g2, const float* be2,
                      float* __restrict__ out) {
  extern __shared__ char smem[];
  float*  Xf  = (float*)(smem + OFF_XF);
  __bf16* Xbf = (__bf16*)(smem + OFF_XBF);
  __bf16* Qb  = (__bf16*)(smem + OFF_Q);
  __bf16* Kb  = (__bf16*)(smem + OFF_K);
  __bf16* Vt  = (__bf16*)(smem + OFF_VT);
  __bf16* Pb  = (__bf16*)(smem + OFF_P);
  __bf16* Ab  = (__bf16*)(smem + OFF_AO);
  __bf16* Wt  = (__bf16*)(smem + OFF_WT);
  __bf16* W1t = (__bf16*)(smem + OFF_Q);   // FF alias (256*136*2 = 69632)
  __bf16* Hb  = (__bf16*)(smem + OFF_VT);  // FF alias (128*264*2 = 67584)

  const int tid  = threadIdx.x;
  const int wave = tid >> 5;
  const int lane = tid & 31;
  const int ln   = lane & 15;
  const int lh   = lane >> 4;
  const long gbase = (long)blockIdx.x * (TOK * MDIM);

  // ---------------- Phase 0: load route slab ----------------
  for (int i = tid; i < TOK * MDIM; i += WGSZ) {
    float v = route[gbase + i];
    Xf[i] = v;
    Xbf[(i >> 7) * LDH + (i & 127)] = (__bf16)v;
  }
  __syncthreads();

  // ---------------- Phase A: Q,K,V projections (fully unrolled: 3 straight-line blocks) ----
  stageT128(wq, Wt, tid);
  __syncthreads();
  proj_gemm<LDH, 1>(Xbf, Wt, bq, Qb, wave, ln, lh);   // Q token-major
  __syncthreads();
  stageT128(wk, Wt, tid);
  __syncthreads();
  proj_gemm<LDH, 1>(Xbf, Wt, bk, Kb, wave, ln, lh);   // K token-major
  __syncthreads();
  stageT128(wv, Wt, tid);
  __syncthreads();
  proj_gemm<1, LDH>(Xbf, Wt, bv, Vt, wave, ln, lh);   // V feature-major (transposed)
  __syncthreads();

  // ---------------- Phase B: attention, wave = head ----------------
  {
    const int h = wave;
    __bf16* Pw = Pb + h * (16 * LDH);
#pragma unroll 1
    for (int qb = 0; qb < 8; ++qb) {
      // A = Q rows (16 queries) x d, zero-padded d16..31
      BF16x16 aq;
      aq.h[0] = *(const v8bf*)(&Qb[(qb * 16 + ln) * LDH + h * HD + lh * 8]);
      aq.h[1] = zero8();
      v8f S[8];
#pragma unroll 1
      for (int nb = 0; nb < 8; ++nb) {
        BF16x16 bkm;
        bkm.h[0] = *(const v8bf*)(&Kb[(nb * 16 + ln) * LDH + h * HD + lh * 8]);
        bkm.h[1] = zero8();
        v8f c;
#pragma unroll
        for (int i = 0; i < 8; ++i) c[i] = 0.f;
        S[nb] = wmma_bf16(aq.v, bkm.v, c);
      }
      // softmax over keys (8 tiles x 16 lanes per row)
#pragma unroll
      for (int v = 0; v < 8; ++v) {
        float mx = -3.4e38f;
#pragma unroll
        for (int nb = 0; nb < 8; ++nb) { S[nb][v] *= 0.25f; mx = fmaxf(mx, S[nb][v]); }
        mx = rmax16(mx);
        float sum = 0.f;
#pragma unroll
        for (int nb = 0; nb < 8; ++nb) { float e = __expf(S[nb][v] - mx); S[nb][v] = e; sum += e; }
        sum = rsum16(sum);
        float inv = 1.f / sum;
#pragma unroll
        for (int nb = 0; nb < 8; ++nb) S[nb][v] *= inv;
      }
      // round-trip probs through wave-private LDS to get A layout
#pragma unroll
      for (int nb = 0; nb < 8; ++nb)
#pragma unroll
        for (int v = 0; v < 8; ++v)
          Pw[(v + 8 * lh) * LDH + nb * 16 + ln] = (__bf16)S[nb][v];
      // out_h = P (16x128) @ V_h (128x16)
      v8f o;
#pragma unroll
      for (int i = 0; i < 8; ++i) o[i] = 0.f;
#pragma unroll
      for (int kk = 0; kk < 4; ++kk) {
        v16bf ap = load_a(&Pw[ln * LDH], kk * 32, lh);
        v16bf bvv = load_b(&Vt[(h * HD + ln) * LDH], kk * 32, lh);
        o = wmma_bf16(ap, bvv, o);
      }
#pragma unroll
      for (int v = 0; v < 8; ++v)
        Ab[(qb * 16 + v + 8 * lh) * LDH + h * HD + ln] = (__bf16)o[v];
    }
  }
  __syncthreads();

  // ---------------- Phase C: Wo projection + residual + LN1 ----------------
  stageT128(wo, Wt, tid);
  __syncthreads();
  {
    v16bf At[4];
#pragma unroll
    for (int kk = 0; kk < 4; ++kk)
      At[kk] = load_a(&Ab[(wave * 16 + ln) * LDH], kk * 32, lh);
    v8f r[8];
#pragma unroll 1
    for (int nb = 0; nb < 8; ++nb) {
      float bias = bo[nb * 16 + ln];
      v8f acc;
#pragma unroll
      for (int i = 0; i < 8; ++i) acc[i] = bias;
      const __bf16* bptr = &Wt[(nb * 16 + ln) * LDH];
#pragma unroll
      for (int kk = 0; kk < 4; ++kk)
        acc = wmma_bf16(At[kk], load_b(bptr, kk * 32, lh), acc);
#pragma unroll
      for (int v = 0; v < 8; ++v)
        acc[v] += Xf[(wave * 16 + v + 8 * lh) * MDIM + nb * 16 + ln];
      r[nb] = acc;
    }
    float mu[8], rs[8];
#pragma unroll
    for (int v = 0; v < 8; ++v) {
      float s = 0.f;
#pragma unroll
      for (int nb = 0; nb < 8; ++nb) s += r[nb][v];
      mu[v] = rsum16(s) * (1.0f / 128.0f);
      float q = 0.f;
#pragma unroll
      for (int nb = 0; nb < 8; ++nb) { float d = r[nb][v] - mu[v]; q += d * d; }
      rs[v] = rsqrtf(rsum16(q) * (1.0f / 128.0f) + 1e-5f);
    }
#pragma unroll
    for (int nb = 0; nb < 8; ++nb) {
      float g = g1[nb * 16 + ln], b = be1[nb * 16 + ln];
#pragma unroll
      for (int v = 0; v < 8; ++v) {
        float xv = (r[nb][v] - mu[v]) * rs[v] * g + b;
        int tok = wave * 16 + v + 8 * lh;
        Xf[tok * MDIM + nb * 16 + ln] = xv;
        Xbf[tok * LDH + nb * 16 + ln] = (__bf16)xv;
      }
    }
  }

  // ---------------- Phase D: FFN in two 256-wide halves ----------------
  v8f yacc[8];
#pragma unroll
  for (int nb = 0; nb < 8; ++nb) {
    float bias = b2[nb * 16 + ln];
#pragma unroll
    for (int i = 0; i < 8; ++i) yacc[nb][i] = bias;
  }
#pragma unroll 1
  for (int half = 0; half < 2; ++half) {
    __syncthreads();
    for (int i = tid; i < 256 * 128; i += WGSZ) {  // stage w1 half, transposed
      int n = i & 255, k = i >> 8;
      W1t[n * LDH + k] = (__bf16)w1[k * FFD + half * 256 + n];
    }
    __syncthreads();
    {
      v16bf At[4];
#pragma unroll
      for (int kk = 0; kk < 4; ++kk)
        At[kk] = load_a(&Xbf[(wave * 16 + ln) * LDH], kk * 32, lh);
#pragma unroll 1
      for (int nb = 0; nb < 16; ++nb) {
        float bias = b1[half * 256 + nb * 16 + ln];
        v8f acc;
#pragma unroll
        for (int i = 0; i < 8; ++i) acc[i] = bias;
        const __bf16* bptr = &W1t[(nb * 16 + ln) * LDH];
#pragma unroll
        for (int kk = 0; kk < 4; ++kk)
          acc = wmma_bf16(At[kk], load_b(bptr, kk * 32, lh), acc);
#pragma unroll
        for (int v = 0; v < 8; ++v)
          Hb[(wave * 16 + v + 8 * lh) * LDH2 + nb * 16 + ln] = (__bf16)fmaxf(acc[v], 0.f);
      }
    }
    __syncthreads();
    __bf16* W2t = W1t;  // alias: w1T dead now
    for (int i = tid; i < 128 * 256; i += WGSZ) {  // stage w2 half, transposed
      int n = i & 127, k = i >> 7;
      W2t[n * LDH2 + k] = (__bf16)w2[(half * 256 + k) * MDIM + n];
    }
    __syncthreads();
#pragma unroll 1
    for (int kk = 0; kk < 8; ++kk) {
      v16bf ah = load_a(&Hb[(wave * 16 + ln) * LDH2], kk * 32, lh);
#pragma unroll
      for (int nb = 0; nb < 8; ++nb)
        yacc[nb] = wmma_bf16(ah, load_b(&W2t[(nb * 16 + ln) * LDH2], kk * 32, lh), yacc[nb]);
    }
  }

  // ---------------- LN2 + store ----------------
  {
    v8f r[8];
#pragma unroll
    for (int nb = 0; nb < 8; ++nb)
#pragma unroll
      for (int v = 0; v < 8; ++v)
        r[nb][v] = yacc[nb][v] + Xf[(wave * 16 + v + 8 * lh) * MDIM + nb * 16 + ln];
    float mu[8], rs[8];
#pragma unroll
    for (int v = 0; v < 8; ++v) {
      float s = 0.f;
#pragma unroll
      for (int nb = 0; nb < 8; ++nb) s += r[nb][v];
      mu[v] = rsum16(s) * (1.0f / 128.0f);
      float q = 0.f;
#pragma unroll
      for (int nb = 0; nb < 8; ++nb) { float d = r[nb][v] - mu[v]; q += d * d; }
      rs[v] = rsqrtf(rsum16(q) * (1.0f / 128.0f) + 1e-5f);
    }
#pragma unroll
    for (int nb = 0; nb < 8; ++nb) {
      float g = g2[nb * 16 + ln], b = be2[nb * 16 + ln];
#pragma unroll
      for (int v = 0; v < 8; ++v) {
        float xv = (r[nb][v] - mu[v]) * rs[v] * g + b;
        out[gbase + (long)((wave * 16 + v + 8 * lh) * MDIM + nb * 16 + ln)] = xv;
      }
    }
  }
}

extern "C" void kernel_launch(void* const* d_in, const int* in_sizes, int n_in,
                              void* d_out, int out_size, void* d_ws, size_t ws_size,
                              hipStream_t stream) {
  (void)in_sizes; (void)n_in; (void)d_ws; (void)ws_size; (void)out_size;
  const float* route = (const float*)d_in[0];
  const float* wq = (const float*)d_in[1];
  const float* bq = (const float*)d_in[2];
  const float* wk = (const float*)d_in[3];
  const float* bk = (const float*)d_in[4];
  const float* wv = (const float*)d_in[5];
  const float* bv = (const float*)d_in[6];
  const float* wo = (const float*)d_in[7];
  const float* bo = (const float*)d_in[8];
  const float* w1 = (const float*)d_in[9];
  const float* b1 = (const float*)d_in[10];
  const float* w2 = (const float*)d_in[11];
  const float* b2 = (const float*)d_in[12];
  const float* g1 = (const float*)d_in[13];
  const float* be1 = (const float*)d_in[14];
  const float* g2 = (const float*)d_in[15];
  const float* be2 = (const float*)d_in[16];

  (void)hipFuncSetAttribute((const void*)fused_node_attn_layer,
                            hipFuncAttributeMaxDynamicSharedMemorySize, SMEM_TOT);
  fused_node_attn_layer<<<dim3(16 * 64), dim3(WGSZ), SMEM_TOT, stream>>>(
      route, wq, bq, wk, bk, wv, bv, wo, bo, w1, b1, w2, b2,
      g1, be1, g2, be2, (float*)d_out);
}
